// KANLinear_21045339750835
// MI455X (gfx1250) — matrix operand order repaired and checked by reading, not compiled, not run on hardware
//
#include <hip/hip_runtime.h>

// KANLinear forward for MI455X (gfx1250, wave32, WMMA).
//   out[b,o] = silu(x) @ Wb^T  +  einsum('bfi,ofi->bo', bspline3(x), Wsp*scaler)
// Both terms run through v_wmma_f32_16x16x32_f16 (f32 accum).
// GEMM uses the CDNA5 async path: global_load_async_to_lds_b128 + ASYNCcnt,
// double-buffered LDS, single barrier per K-step.
//
// Workspace layout (f16):
//   A1 [4096,1024]  = silu(x)
//   A2 [4096,8192]  = basis, K index = f*8+i
//   W1 [1024,1024]  = base_weight            (already [out,K] layout)
//   W2 [1024,8192]  = spline_weight*scaler   (already [out,K] layout)
// total ~90 MB.

#define BATCH 4096
#define IN_F  1024
#define OUT_F 1024
#define NSPL  8
#define K2    (IN_F * NSPL) // 8192

typedef _Float16 half16  __attribute__((ext_vector_type(16)));
typedef _Float16 half8   __attribute__((ext_vector_type(8)));
typedef float    floatx8 __attribute__((ext_vector_type(8)));

// ---------------------------------------------------------------- prep: x ---
__global__ __launch_bounds__(256)
void kan_prep_x(const float* __restrict__ x, const float* __restrict__ grid,
                _Float16* __restrict__ A1, _Float16* __restrict__ A2) {
  int idx = blockIdx.x * 256 + threadIdx.x;
  if (idx >= BATCH * IN_F) return;

  // all grid rows are identical -> read row 0 (uniform -> scalar loads)
  float g[12];
#pragma unroll
  for (int j = 0; j < 12; ++j) g[j] = grid[j];

  float xv = x[idx];
  float sig = 1.0f / (1.0f + __expf(-xv));
  A1[idx] = (_Float16)(xv * sig); // silu

  // cubic B-spline basis (Cox-de Boor); uniform grid => denom = k*h
  float b[11];
#pragma unroll
  for (int j = 0; j < 11; ++j)
    b[j] = (xv >= g[j] && xv < g[j + 1]) ? 1.0f : 0.0f;
#pragma unroll
  for (int k = 1; k <= 3; ++k) {
    float invkh = 1.0f / (g[k] - g[0]); // == 1/(k*h)
#pragma unroll
    for (int j = 0; j <= 10 - k; ++j) {
      float left  = (xv - g[j]) * invkh;
      float right = (g[j + k + 1] - xv) * invkh;
      b[j] = left * b[j] + right * b[j + 1];
    }
  }
  half8 o;
#pragma unroll
  for (int i = 0; i < 8; ++i) o[i] = (_Float16)b[i];
  *(half8*)(A2 + (size_t)idx * NSPL) = o; // 16B coalesced store
}

// ---------------------------------------------------------- prep: weights ---
__global__ __launch_bounds__(256)
void kan_prep_w(const float* __restrict__ bw, const float* __restrict__ sw,
                const float* __restrict__ sc,
                _Float16* __restrict__ W1, _Float16* __restrict__ W2) {
  int idx = blockIdx.x * 256 + threadIdx.x;
  if (idx < OUT_F * IN_F) W1[idx] = (_Float16)bw[idx];
  if (idx < OUT_F * IN_F * NSPL)
    W2[idx] = (_Float16)(sw[idx] * sc[idx >> 3]); // idx>>3 == (o*IN_F + f)
}

// ------------------------------------------------------------------- GEMM ---
#define BM 128
#define BN 128
#define BK 32
#define LDS_STRIDE 40 // 32 halves + 8 pad; 80B rows, 16B-aligned, conflict-free
#define TILE_HALVES (BM * LDS_STRIDE) // 5120 halves
#define TILE_BYTES  (TILE_HALVES * 2) // 10240 B
// smem: [buf0 A][buf0 B][buf1 A][buf1 B] = 40960 B

// CDNA5 async copy: memory -> LDS directly, tracked by ASYNCcnt.
#define ASYNC_LOAD_B128(ldsaddr, gaddr)                          \
  asm volatile("global_load_async_to_lds_b128 %0, %1, off"       \
               :: "v"(ldsaddr), "v"(gaddr) : "memory")
#define WAIT_ASYNCCNT_0() asm volatile("s_wait_asynccnt 0x0" ::: "memory")

__device__ __forceinline__ void pack16(half16& d, half8 lo, half8 hi) {
#pragma unroll
  for (int i = 0; i < 8; ++i) { d[i] = lo[i]; d[i + 8] = hi[i]; }
}

// Issue one 128x32 A-tile + 128x32 B-tile as per-lane async b128 copies.
__device__ __forceinline__ void issue_tile_async(
    const _Float16* __restrict__ Ag, const _Float16* __restrict__ Bg,
    int K, int k0, unsigned ldsA, unsigned ldsB,
    int r0, int o0, int r1, int o1) {
  unsigned long long a0 = (unsigned long long)(Ag + (size_t)r0 * K + k0 + o0);
  unsigned long long a1 = (unsigned long long)(Ag + (size_t)r1 * K + k0 + o1);
  unsigned long long b0 = (unsigned long long)(Bg + (size_t)r0 * K + k0 + o0);
  unsigned long long b1 = (unsigned long long)(Bg + (size_t)r1 * K + k0 + o1);
  unsigned dA0 = ldsA + (unsigned)(r0 * LDS_STRIDE + o0) * 2u;
  unsigned dA1 = ldsA + (unsigned)(r1 * LDS_STRIDE + o1) * 2u;
  unsigned dB0 = ldsB + (unsigned)(r0 * LDS_STRIDE + o0) * 2u;
  unsigned dB1 = ldsB + (unsigned)(r1 * LDS_STRIDE + o1) * 2u;
  ASYNC_LOAD_B128(dA0, a0);
  ASYNC_LOAD_B128(dA1, a1);
  ASYNC_LOAD_B128(dB0, b0);
  ASYNC_LOAD_B128(dB1, b1);
}

// 8 WMMAs on one staged tile pair (wave owns 32(M) x 64(N)).
__device__ __forceinline__ void mma_tile(
    const _Float16* __restrict__ As, const _Float16* __restrict__ Bs,
    int lane, int waveM, int waveN, floatx8 (&acc)[2][4]) {
  const int krow = lane >> 4; // half-wave select
  const int lrow = lane & 15;
  // B fragment (32x16 f16): lane holds col N=lrow, K = krow*16 .. +15
  half16 bf[4];
#pragma unroll
  for (int n = 0; n < 4; ++n) {
    const _Float16* bp =
        Bs + (waveN * 64 + n * 16 + lrow) * LDS_STRIDE + krow * 16;
    pack16(bf[n], *(const half8*)bp, *(const half8*)(bp + 8));
  }
  // A fragment (16x32 f16): lane holds row M=lrow, K = krow*16 .. +15
#pragma unroll
  for (int m = 0; m < 2; ++m) {
    const _Float16* ap = As + (waveM * 32 + m * 16 + lrow) * LDS_STRIDE;
    half16 af;
    pack16(af, *(const half8*)(ap + krow * 16),
               *(const half8*)(ap + krow * 16 + 8));
#pragma unroll
    for (int n = 0; n < 4; ++n)
      acc[m][n] = __builtin_amdgcn_wmma_f32_16x16x32_f16(
          false, af, false, bf[n], (short)0, acc[m][n], false, false);
  }
}

// Double-buffered async K-loop; one barrier per K-step.
// Safety: issuing into buf[1-cur] after barrier(it) is race-free because
// barrier(it) proves every wave finished mma(it-1) reads of buf[1-cur] and
// drained (s_wait_asynccnt 0) its own writes of buf[cur].
__device__ __forceinline__ void gemm_k_async(
    const _Float16* __restrict__ Ag, const _Float16* __restrict__ Bg, int K,
    const _Float16* smem, unsigned lds_base,
    int tid, int lane, int waveM, int waveN, floatx8 (&acc)[2][4]) {
  const int c0 = tid, c1 = tid + 256; // 512 16B-chunks per 128x32 tile
  const int r0 = c0 >> 2, o0 = (c0 & 3) * 8;
  const int r1 = c1 >> 2, o1 = (c1 & 3) * 8;
  const int niter = K / BK;

  issue_tile_async(Ag, Bg, K, 0, lds_base, lds_base + TILE_BYTES,
                   r0, o0, r1, o1);
  for (int it = 0; it < niter; ++it) {
    const int cur = it & 1;
    WAIT_ASYNCCNT_0(); // my async writes of buf[cur] are in LDS
    __syncthreads();   // everyone's writes done; everyone done reading other buf
    if (it + 1 < niter) {
      unsigned nb = lds_base + (unsigned)(1 - cur) * (2u * TILE_BYTES);
      issue_tile_async(Ag, Bg, K, (it + 1) * BK, nb, nb + TILE_BYTES,
                       r0, o0, r1, o1); // overlaps the WMMAs below
    }
    const _Float16* As = smem + (size_t)cur * (2 * TILE_HALVES);
    const _Float16* Bs = As + TILE_HALVES;
    mma_tile(As, Bs, lane, waveM, waveN, acc);
  }
}

__global__ __launch_bounds__(256)
void kan_gemm(const _Float16* __restrict__ A1, const _Float16* __restrict__ A2,
              const _Float16* __restrict__ W1, const _Float16* __restrict__ W2,
              float* __restrict__ C) {
  __shared__ _Float16 smem[4 * TILE_HALVES]; // 40960 B: A0,B0,A1,B1
  const unsigned lds_base = (unsigned)(unsigned long long)&smem[0];
  const int tid = threadIdx.x;
  const int lane = tid & 31;
  const int wv = tid >> 5;   // 8 waves
  const int waveM = wv & 3;  // 4 over M (32 rows each)
  const int waveN = wv >> 2; // 2 over N (64 cols each)
  const int bM = blockIdx.x * BM;
  const int bN = blockIdx.y * BN;

  floatx8 acc[2][4];
#pragma unroll
  for (int m = 0; m < 2; ++m)
#pragma unroll
    for (int n = 0; n < 4; ++n)
#pragma unroll
      for (int v = 0; v < 8; ++v) acc[m][n][v] = 0.0f;

  // base term: K = 1024 (32 K-steps)
  gemm_k_async(A1 + (size_t)bM * IN_F, W1 + (size_t)bN * IN_F, IN_F,
               smem, lds_base, tid, lane, waveM, waveN, acc);
  // spline term: K = 8192 (256 K-steps), same accumulators
  gemm_k_async(A2 + (size_t)bM * K2, W2 + (size_t)bN * K2, K2,
               smem, lds_base, tid, lane, waveM, waveN, acc);

  // C layout: VGPR v -> M = (lane>>4)*8 + v, N = lane&15
  const int krow = lane >> 4;
  const int lrow = lane & 15;
#pragma unroll
  for (int m = 0; m < 2; ++m) {
#pragma unroll
    for (int n = 0; n < 4; ++n) {
      const int row0 = bM + waveM * 32 + m * 16 + krow * 8;
      const int col  = bN + waveN * 64 + n * 16 + lrow;
#pragma unroll
      for (int v = 0; v < 8; ++v)
        C[(size_t)(row0 + v) * OUT_F + col] = acc[m][n][v];
    }
  }
}

// ----------------------------------------------------------------- launch ---
extern "C" void kernel_launch(void* const* d_in, const int* in_sizes, int n_in,
                              void* d_out, int out_size, void* d_ws, size_t ws_size,
                              hipStream_t stream) {
  const float* x    = (const float*)d_in[0]; // [4096,1024]
  const float* bw   = (const float*)d_in[1]; // [1024,1024]
  const float* sw   = (const float*)d_in[2]; // [1024,1024,8]
  const float* sc   = (const float*)d_in[3]; // [1024,1024]
  const float* grid = (const float*)d_in[4]; // [1024,12]
  float* out = (float*)d_out;                // [4096,1024] f32

  _Float16* A1 = (_Float16*)d_ws;
  _Float16* A2 = A1 + (size_t)BATCH * IN_F;
  _Float16* W1 = A2 + (size_t)BATCH * K2;
  _Float16* W2 = W1 + (size_t)OUT_F * IN_F;
  // requires ws_size >= 2*(BATCH*9216 + OUT_F*9216) ~ 90 MB

  kan_prep_x<<<(BATCH * IN_F) / 256, 256, 0, stream>>>(x, grid, A1, A2);
  kan_prep_w<<<(OUT_F * IN_F * NSPL) / 256, 256, 0, stream>>>(bw, sw, sc, W1, W2);
  dim3 g(BATCH / BM, OUT_F / BN); // 32 x 8 blocks
  kan_gemm<<<g, 256, 0, stream>>>(A1, A2, W1, W2, out);
}
